// MLP4D_7249904796406
// MI455X (gfx1250) — matrix operand order repaired and checked by reading, not compiled, not run on hardware
//
#include <hip/hip_runtime.h>
#include <hip/hip_bf16.h>
#include <cmath>

typedef __attribute__((ext_vector_type(16))) _Float16 v16h;
typedef __attribute__((ext_vector_type(8)))  _Float16 v8h;
typedef __attribute__((ext_vector_type(2)))  _Float16 v2h;
typedef __attribute__((ext_vector_type(8)))  float    v8f;

// ---------------------------------------------------------------------------
// Geometry (hardcoded from the reference problem)
// ---------------------------------------------------------------------------
#define BN   8
#define HH   256
#define WW   256
#define TN   128             // pixels per block (half of one image row)
#define NT   (TN / 16)       // 8 N-tiles of 16 pixels

// ---------------------------------------------------------------------------
// Branch-free exact-GELU: Abramowitz-Stegun 7.1.26 erf, |err| <= 1.5e-7.
// ~15 VALU ops incl. one v_rcp + one v_exp (TRANS, co-executes). No EXEC
// divergence, unlike ocml erff.
// ---------------------------------------------------------------------------
__device__ __forceinline__ float gelu_exact(float y) {
    float z = __builtin_fabsf(y) * 0.70710678118654752f;
    float t = __builtin_amdgcn_rcpf(__builtin_fmaf(0.3275911f, z, 1.0f));
    float p = t * __builtin_fmaf(t, __builtin_fmaf(t, __builtin_fmaf(t,
                  __builtin_fmaf(t, 1.061405429f, -1.453152027f),
                  1.421413741f), -0.284496736f), 0.254829592f);
    float e = __builtin_amdgcn_exp2f(-z * z * 1.4426950408889634f);
    float erf_abs = __builtin_fmaf(-p, e, 1.0f);
    float s = __builtin_copysignf(erf_abs, y);
    return 0.5f * y * (1.0f + s);
}

// ---------------------------------------------------------------------------
// Prep: pack both weight matrices into per-lane WMMA A-fragment order and
// zero-pad both biases to MT*16. A-fragment k order used by this code (contiguous, matched by B):
//   lane 0-15 : M=lane,    halves 0-15 -> K=0..15
//   lane16-31 : M=lane-16, halves 0-15 -> K=16..31
// ---------------------------------------------------------------------------
__device__ __forceinline__ void pack_one(const float* __restrict__ w,
                                         _Float16* __restrict__ wp,
                                         int idx, int O, int C, int KS) {
    int half = idx & 15;
    int lane = (idx >> 4) & 31;
    int ks   = (idx >> 9) % KS;
    int mt   = idx / (KS * 512);
    int m = mt * 16 + (lane & 15);
    int k = ks * 32 + ((lane >> 4) << 4) + half;
    float v = (m < O && k < C) ? w[m * C + k] : 0.0f;
    wp[idx] = (_Float16)v;
}

#define T1 (13 * 4 * 512)    // w1p elements
#define T2 (7 * 7 * 512)     // w2p elements

__global__ void prep_kernel(const float* __restrict__ w1, const float* __restrict__ b1,
                            const float* __restrict__ w2, const float* __restrict__ b2,
                            _Float16* __restrict__ w1p, _Float16* __restrict__ w2p,
                            float* __restrict__ b1p, float* __restrict__ b2p) {
    int idx = blockIdx.x * blockDim.x + threadIdx.x;
    if (idx < T1) {
        pack_one(w1, w1p, idx, 200, 100, 4);
    } else if (idx < T1 + T2) {
        pack_one(w2, w2p, idx - T1, 100, 200, 7);
    } else if (idx < T1 + T2 + 208) {
        int i = idx - (T1 + T2);
        b1p[i] = (i < 200) ? b1[i] : 0.0f;
    } else if (idx < T1 + T2 + 208 + 112) {
        int i = idx - (T1 + T2 + 208);
        b2p[i] = (i < 100) ? b2[i] : 0.0f;
    }
}

// ---------------------------------------------------------------------------
// Fused shift + 1x1 conv (+bias, optional exact GELU) as WMMA GEMM.
// A wave processes its M-tiles in separate passes: each pass holds exactly
// one A-fragment set (KS x 8 VGPRs) + one bias vector in registers, with a
// rolled nt loop (unroll 2) -> no rematerialization of A from global inside
// the hot loop. B fragments come from LDS split lo/hi 16B arrays (per-lane
// stride 4 dwords -> conflict-free half-wave ds_load_b128).
// ---------------------------------------------------------------------------
template <int C, int CSTR, int KP, int CO, int COP, int MT, bool DO_GELU,
          typename TIN, typename TOUT>
__global__ __launch_bounds__(256)
void shift_gemm_kernel(const TIN* __restrict__ in,
                       const _Float16* __restrict__ wp,
                       const float* __restrict__ biasp,   // padded to MT*16
                       TOUT* __restrict__ out) {
    constexpr int KS  = KP / 32;
    constexpr int CGP = C / 5;
    constexpr int NF  = NT * KS;

    __shared__ __align__(16) _Float16 xs_lo[NF * 32 * 8];
    __shared__ __align__(16) _Float16 xs_hi[NF * 32 * 8];

    const int tid  = threadIdx.x;
    const int bid  = blockIdx.x;             // 0 .. BN*HH*2-1
    const int b    = bid >> 9;
    const int rem  = bid & 511;
    const int hrow = rem >> 1;
    const int w0   = (rem & 1) << 7;

    // ---- stage shifted input into LDS; thread stages 2 adjacent channels ----
    {
        const int n    = tid & 127;
        const int csub = (tid >> 7) << 1;    // 0 or 2
        #pragma unroll
        for (int g = 0; g < 5; ++g) {        // shift group: all control hoisted
            const int dh = (g == 2) ? 1 : (g == 3) ? -1 : 0;
            const int dw = (g == 0) ? 1 : (g == 1) ? -1 : 0;
            const int r   = hrow + dh;
            const int col = w0 + n + dw;
            const bool ok = ((unsigned)r < HH) && ((unsigned)col < WW);
            const long long rowbase =
                (((long long)b * CSTR + g * CGP) * HH + r) * WW + col;
            #pragma unroll
            for (int itg = 0; itg < CGP / 4; ++itg) {
                const int cp = g * CGP + itg * 4 + csub;     // even channel
                _Float16 h0 = (_Float16)0.0f, h1 = (_Float16)0.0f;
                if (ok) {
                    long long a = rowbase + (long long)(itg * 4 + csub) * (HH * WW);
                    h0 = (_Float16)in[a];
                    h1 = (_Float16)in[a + HH * WW];
                }
                const int F     = (n >> 4) * KS + (cp >> 5);
                const int lanef = (n & 15) | (((cp >> 4) & 1) << 4);
                const int half  = cp & 15;
                v2h pv = { h0, h1 };
                if (half < 8)
                    *(v2h*)&xs_lo[(F * 32 + lanef) * 8 + half] = pv;
                else
                    *(v2h*)&xs_hi[(F * 32 + lanef) * 8 + (half - 8)] = pv;
            }
        }
        // zero-fill K padding channels
        #pragma unroll
        for (int cp0 = C + csub; cp0 < KP; cp0 += 4) {
            const int F     = (n >> 4) * KS + (cp0 >> 5);
            const int lanef = (n & 15) | (((cp0 >> 4) & 1) << 4);
            const int half  = cp0 & 15;
            v2h pv = { (_Float16)0.0f, (_Float16)0.0f };
            if (half < 8)
                *(v2h*)&xs_lo[(F * 32 + lanef) * 8 + half] = pv;
            else
                *(v2h*)&xs_hi[(F * 32 + lanef) * 8 + (half - 8)] = pv;
        }
    }
    __syncthreads();

    // ---- WMMA compute: one M-tile per pass, A + bias register-resident ----
    const int lane  = tid & 31;
    const int wv    = tid >> 5;
    const int mhalf = (lane >> 4) << 3;

    for (int mt = wv; mt < MT; mt += 8) {        // wave-uniform, 1-2 passes
        v16h a[KS];
        #pragma unroll
        for (int ks = 0; ks < KS; ++ks)
            a[ks] = *(const v16h*)(wp + ((size_t)(mt * KS + ks) * 512 + lane * 16));

        // C/D layout: lane L -> N = L%16, M = mt*16 + (L/16)*8 + r
        const int mbase = mt * 16 + mhalf;
        const v8f bv   = *(const v8f*)(biasp + mt * 16 + mhalf);
        const bool full = (mt * 16 + 16 <= COP);  // uniform, hoisted

        #pragma unroll 2
        for (int nt = 0; nt < NT; ++nt) {
            v16h bfr[KS];
            #pragma unroll
            for (int ks = 0; ks < KS; ++ks) {
                const int F = nt * KS + ks;
                v8h lo = *(const v8h*)&xs_lo[(F * 32 + lane) * 8];
                v8h hi = *(const v8h*)&xs_hi[(F * 32 + lane) * 8];
                bfr[ks] = __builtin_shufflevector(lo, hi,
                    0, 1, 2, 3, 4, 5, 6, 7, 8, 9, 10, 11, 12, 13, 14, 15);
            }

            v8f acc = {};
            #pragma unroll
            for (int ks = 0; ks < KS; ++ks)
                acc = __builtin_amdgcn_wmma_f32_16x16x32_f16(
                    false, a[ks], false, bfr[ks], (short)0, acc, false, false);

            const int col = w0 + nt * 16 + (lane & 15);
            const size_t obase = (((size_t)b * COP + mbase) * HH + hrow) * WW + col;
            if (full) {                          // unconditional stores
                #pragma unroll
                for (int rr = 0; rr < 8; ++rr) {
                    float y = acc[rr] + bv[rr];
                    if (DO_GELU) {
                        out[obase + (size_t)rr * (HH * WW)] = (TOUT)gelu_exact(y);
                    } else {
                        __builtin_nontemporal_store((TOUT)y,
                            &out[obase + (size_t)rr * (HH * WW)]);
                    }
                }
            } else {                              // tail M-tile only
                #pragma unroll
                for (int rr = 0; rr < 8; ++rr) {
                    if (mbase + rr < CO) {
                        float y = acc[rr] + bv[rr];
                        if (DO_GELU) y = gelu_exact(y);
                        out[obase + (size_t)rr * (HH * WW)] = (TOUT)y;
                    }
                }
            }
        }
    }
}

// ---------------------------------------------------------------------------
// Launch: prep -> layer1 (f32 in, padded f16 h out, gelu) -> layer2 (f32 out)
// ---------------------------------------------------------------------------
extern "C" void kernel_launch(void* const* d_in, const int* in_sizes, int n_in,
                              void* d_out, int out_size, void* d_ws, size_t ws_size,
                              hipStream_t stream) {
    const float* x  = (const float*)d_in[0];   // [8,100,256,256]
    const float* w1 = (const float*)d_in[1];   // [200,100]
    const float* b1 = (const float*)d_in[2];   // [200]
    const float* w2 = (const float*)d_in[3];   // [100,200]
    const float* b2 = (const float*)d_in[4];   // [100]
    float* out = (float*)d_out;                // [8,100,256,256]

    char* ws = (char*)d_ws;
    const size_t hbytes = (size_t)BN * 208 * HH * WW * sizeof(_Float16); // padded h
    _Float16* hbuf = (_Float16*)ws;
    _Float16* w1p  = (_Float16*)(ws + hbytes);
    _Float16* w2p  = (_Float16*)(ws + hbytes + (size_t)T1 * 2);
    float*    b1p  = (float*)   (ws + hbytes + (size_t)T1 * 2 + (size_t)T2 * 2);
    float*    b2p  = b1p + 208;

    {
        int total = T1 + T2 + 208 + 112;
        prep_kernel<<<(total + 255) / 256, 256, 0, stream>>>(
            w1, b1, w2, b2, w1p, w2p, b1p, b2p);
    }

    const int nblk = BN * HH * (WW / TN);      // 4096
    // Layer 1: C=100 (stride 100), K=128, CO=200 stored padded to 208, GELU
    shift_gemm_kernel<100, 100, 128, 200, 208, 13, true,  float,    _Float16>
        <<<nblk, 256, 0, stream>>>(x, w1p, b1p, hbuf);
    // Layer 2: C=200 (stride 208 in padded h), K=224, CO=100, f32 out
    shift_gemm_kernel<200, 208, 224, 100, 100, 7,  false, _Float16, float>
        <<<nblk, 256, 0, stream>>>(hbuf, w2p, b2p, out);
}
